// PredictorCell_63556926046386
// MI455X (gfx1250) — compile-verified, matrix operand order checked by reading, not run on hardware
//
#include <hip/hip_runtime.h>

// Problem dims
#define BATCH 8192
#define I_N   256
#define H_N   512
#define MROWS 32           // batch rows per workgroup (2 WMMA M-tiles)

// WMMA fragment vector types (gfx1250, wave32)
typedef __attribute__((ext_vector_type(16))) __bf16       v16bf;
typedef __attribute__((ext_vector_type(8)))  float        v8f;
typedef __attribute__((ext_vector_type(4)))  unsigned int v4u;

union FragU { v16bf bf; v4u u[2]; };

#define WMMA_BF16(a, b, c) \
  __builtin_amdgcn_wmma_f32_16x16x32_bf16(false, (a), false, (b), (short)0, (c), false, false)

__device__ __forceinline__ unsigned short f2bf(float f) {
  unsigned int u = __builtin_bit_cast(unsigned int, f);
  u += 0x7FFFu + ((u >> 16) & 1u);          // round-to-nearest-even
  return (unsigned short)(u >> 16);
}
__device__ __forceinline__ float bf2f(unsigned short s) {
  return __builtin_bit_cast(float, ((unsigned int)s) << 16);
}
__device__ __forceinline__ float sigmoid_f(float x) { return 1.0f / (1.0f + __expf(-x)); }

// ---- A-fragment (16x32 bf16) from LDS, row-major, element stride `stride`.
// ISA 16-bit A layout: lanes 0-15: M=lane, K = k0+{0..7, 16..23};
//                      lanes 16-31: M=lane-16, K = k0+{8..15, 24..31}.
__device__ __forceinline__ v16bf lds_frag(const unsigned short* base, int stride, int k0, int lane) {
  int r  = lane & 15;
  int kb = k0 + ((lane & 16) ? 8 : 0);
  const v4u* q = (const v4u*)(base + r * stride + kb);
  FragU f;
  f.u[0] = q[0];   // K kb .. kb+7
  f.u[1] = q[2];   // K kb+16 .. kb+23
  return f.bf;
}

// ---- B-fragment (32x16 bf16) from global pre-packed weights:
// frag (nb, kk) stored as 32 lanes x 16 bf16, contiguous per lane (fully coalesced).
__device__ __forceinline__ const unsigned short* gbl_frag_ptr(const unsigned short* w, int KT,
                                                              int nb, int kk, int lane) {
  return w + (((size_t)((nb * KT + kk) * 32 + lane)) << 4);
}
__device__ __forceinline__ v16bf gbl_frag(const unsigned short* w, int KT, int nb, int kk, int lane) {
  const v4u* q = (const v4u*)gbl_frag_ptr(w, KT, nb, kk, lane);
  FragU f;
  f.u[0] = q[0];
  f.u[1] = q[1];
  return f.bf;
}

// ============================================================================
// Weight repack: fp32 row-major [N,K] -> bf16 WMMA B-fragment order.
// ============================================================================
__global__ void pack_weights_bf16(const float* __restrict__ src, unsigned short* __restrict__ dst,
                                  int N, int K) {
  int idx = blockIdx.x * blockDim.x + threadIdx.x;
  int total = N * K;
  if (idx >= total) return;
  int j    = idx & 15;
  int lane = (idx >> 4) & 31;
  int fi   = idx >> 9;
  int KT   = K >> 5;
  int kk   = fi % KT;
  int nb   = fi / KT;
  int n    = nb * 16 + (lane & 15);
  int kb   = kk * 32 + ((lane < 16) ? 0 : 8);
  int k    = kb + ((j < 8) ? j : (j + 8));
  dst[idx] = f2bf(src[(size_t)n * K + k]);
}

// ============================================================================
// Fused PredictorCell. 32-row M-block per workgroup (2 WMMA M-tiles sharing
// every B-fragment load), 8 wave32s split the N dimension of each GEMM.
// ============================================================================
#define XH_S   776   // 768 + 8 pad (bf16 elems) -> conflict-free ds_load_b128
#define A0_S   264   // 256 + 8
#define XATT_S 264
#define HN_S   520   // 512 + 8

// Manual LDS layout (bytes); out0 aliases {a0, xatt, logits}, all dead by then.
#define OFF_XH    0
#define SZ_XH     (MROWS * XH_S * 2)              // 49664
#define OFF_HNEW  (OFF_XH + SZ_XH)
#define SZ_HNEW   (MROWS * HN_S * 2)              // 33280
#define OFF_U     (OFF_HNEW + SZ_HNEW)            // 82944
#define OFF_A0    (OFF_U)
#define SZ_A0     (MROWS * A0_S * 2)              // 16896
#define OFF_XATT  (OFF_A0 + SZ_A0)
#define SZ_XATT   (MROWS * XATT_S * 2)            // 16896
#define OFF_LOG   (OFF_XATT + SZ_XATT)
#define SZ_LOG    (MROWS * 256 * 4)               // 32768
#define OFF_OUT0  (OFF_U)                          // 65536 <= 66560 union
#define LDS_TOTAL (OFF_LOG + SZ_LOG)              // 149504 (<160KB -> 2 WG/WGP)

__global__ __launch_bounds__(256) void predictor_cell_kernel(
    const float* __restrict__ x, const float* __restrict__ h, const float* __restrict__ c,
    const float* __restrict__ ab0, const float* __restrict__ ab1,
    const float* __restrict__ b_ih, const float* __restrict__ b_hh,
    const float* __restrict__ ob0, const float* __restrict__ ow1, const float* __restrict__ ob1,
    const unsigned short* __restrict__ aw0p,   // [256,768]  packed, KT=24
    const unsigned short* __restrict__ aw1p,   // [256,256]  packed, KT=8
    const unsigned short* __restrict__ wihp,   // [2048,256] packed, KT=8
    const unsigned short* __restrict__ whhp,   // [2048,512] packed, KT=16
    const unsigned short* __restrict__ ow0p,   // [512,512]  packed, KT=16
    float* __restrict__ o_out, float* __restrict__ o_h,
    float* __restrict__ o_c, float* __restrict__ o_attn) {

  __shared__ __attribute__((aligned(16))) char smem[LDS_TOTAL];
  unsigned short* s_xh   = (unsigned short*)(smem + OFF_XH);
  unsigned short* s_hnew = (unsigned short*)(smem + OFF_HNEW);
  unsigned short* s_a0   = (unsigned short*)(smem + OFF_A0);
  unsigned short* s_xatt = (unsigned short*)(smem + OFF_XATT);
  float*          s_log  = (float*)(smem + OFF_LOG);
  float*          s_out0 = (float*)(smem + OFF_OUT0);

  const int tid  = threadIdx.x;
  const int lane = tid & 31;
  const int wv   = tid >> 5;              // wave id 0..7
  const int m0   = blockIdx.x * MROWS;    // first batch row of this block
  const int n_l  = lane & 15;
  const int mb   = (lane & 16) ? 8 : 0;

  // ---- Stage 0: stage x|h into LDS as bf16 ------------------------------
  for (int idx = tid; idx < MROWS * I_N; idx += 256) {
    int r = idx >> 8, cc = idx & 255;
    s_xh[r * XH_S + cc] = f2bf(x[(size_t)(m0 + r) * I_N + cc]);
  }
  for (int idx = tid; idx < MROWS * H_N; idx += 256) {
    int r = idx >> 9, cc = idx & 511;
    s_xh[r * XH_S + 256 + cc] = f2bf(h[(size_t)(m0 + r) * H_N + cc]);
  }
  __syncthreads();

  // ---- Stage A: a0 = relu(xh @ aw0^T + ab0)   [32 x 256] ----------------
  for (int t = 0; t < 2; ++t) {
    int nb = wv * 2 + t;
    v8f acc0 = {}, acc1 = {};
    for (int kk = 0; kk < 24; ++kk) {
      if (kk + 1 < 24) __builtin_prefetch(gbl_frag_ptr(aw0p, 24, nb, kk + 1, lane), 0, 3);
      v16bf b  = gbl_frag(aw0p, 24, nb, kk, lane);
      v16bf a0 = lds_frag(s_xh,             XH_S, kk * 32, lane);
      v16bf a1 = lds_frag(s_xh + 16 * XH_S, XH_S, kk * 32, lane);
      acc0 = WMMA_BF16(a0, b, acc0);
      acc1 = WMMA_BF16(a1, b, acc1);
    }
    int n = nb * 16 + n_l;
    float bias = ab0[n];
    for (int v = 0; v < 8; ++v) {
      float z0 = acc0[v] + bias, z1 = acc1[v] + bias;
      s_a0[(mb + v) * A0_S + n]      = f2bf(z0 > 0.0f ? z0 : 0.0f);
      s_a0[(16 + mb + v) * A0_S + n] = f2bf(z1 > 0.0f ? z1 : 0.0f);
    }
  }
  __syncthreads();

  // ---- Stage L: logits = a0 @ aw1^T + ab1   [32 x 256] ------------------
  for (int t = 0; t < 2; ++t) {
    int nb = wv * 2 + t;
    v8f acc0 = {}, acc1 = {};
    for (int kk = 0; kk < 8; ++kk) {
      if (kk + 1 < 8) __builtin_prefetch(gbl_frag_ptr(aw1p, 8, nb, kk + 1, lane), 0, 3);
      v16bf b  = gbl_frag(aw1p, 8, nb, kk, lane);
      v16bf a0 = lds_frag(s_a0,             A0_S, kk * 32, lane);
      v16bf a1 = lds_frag(s_a0 + 16 * A0_S, A0_S, kk * 32, lane);
      acc0 = WMMA_BF16(a0, b, acc0);
      acc1 = WMMA_BF16(a1, b, acc1);
    }
    int n = nb * 16 + n_l;
    float bias = ab1[n];
    for (int v = 0; v < 8; ++v) {
      s_log[(mb + v) * 256 + n]      = acc0[v] + bias;
      s_log[(16 + mb + v) * 256 + n] = acc1[v] + bias;
    }
  }
  __syncthreads();

  // ---- Stage S: row softmax -> attn (global) and x*attn (LDS bf16) ------
  for (int rr = 0; rr < 4; ++rr) {
    int r = wv * 4 + rr;
    float vals[8];
    float vmax = -1e30f;
    for (int j = 0; j < 8; ++j) {
      vals[j] = s_log[r * 256 + lane + 32 * j];
      vmax = fmaxf(vmax, vals[j]);
    }
    for (int off = 16; off; off >>= 1) vmax = fmaxf(vmax, __shfl_xor(vmax, off, 32));
    float sum = 0.0f;
    for (int j = 0; j < 8; ++j) { vals[j] = __expf(vals[j] - vmax); sum += vals[j]; }
    for (int off = 16; off; off >>= 1) sum += __shfl_xor(sum, off, 32);
    float inv = 1.0f / sum;
    for (int j = 0; j < 8; ++j) {
      int n = lane + 32 * j;
      float a = vals[j] * inv;
      o_attn[(size_t)(m0 + r) * I_N + n] = a;
      s_xatt[r * XATT_S + n] = f2bf(bf2f(s_xh[r * XH_S + n]) * a);
    }
  }
  __syncthreads();

  // ---- Stage G: gates + LSTM pointwise ----------------------------------
  // wave wv owns hidden columns [wv*64, wv*64+64): 4 tiles x 4 gates x 2 M-tiles.
  for (int t = 0; t < 4; ++t) {
    int nbh = wv * 4 + t;                 // tile index within one gate (0..31)
    v8f aI[2] = {{}, {}}, aF[2] = {{}, {}}, aG[2] = {{}, {}}, aO[2] = {{}, {}};
    for (int kk = 0; kk < 24; ++kk) {
      const unsigned short* src;
      const unsigned short* wb;
      int stride, KT, kw;
      if (kk < 8) { src = s_xatt;      stride = XATT_S; wb = wihp; KT = 8;  kw = kk; }
      else        { src = s_xh + 256;  stride = XH_S;   wb = whhp; KT = 16; kw = kk - 8; }
      // prefetch next k-step's first B fragment (WGP scope)
      if (kk + 1 < 24) {
        const unsigned short* nwb = (kk + 1 < 8) ? wihp : whhp;
        int nKT = (kk + 1 < 8) ? 8 : 16;
        int nkw = (kk + 1 < 8) ? kk + 1 : kk + 1 - 8;
        __builtin_prefetch(gbl_frag_ptr(nwb, nKT, nbh, nkw, lane), 0, 3);
      }
      v16bf a0 = lds_frag(src,               stride, kw * 32, lane);
      v16bf a1 = lds_frag(src + 16 * stride, stride, kw * 32, lane);
      v16bf bI = gbl_frag(wb, KT,      nbh, kw, lane);
      v16bf bF = gbl_frag(wb, KT, 32 + nbh, kw, lane);
      v16bf bG = gbl_frag(wb, KT, 64 + nbh, kw, lane);
      v16bf bO = gbl_frag(wb, KT, 96 + nbh, kw, lane);
      aI[0] = WMMA_BF16(a0, bI, aI[0]);  aI[1] = WMMA_BF16(a1, bI, aI[1]);
      aF[0] = WMMA_BF16(a0, bF, aF[0]);  aF[1] = WMMA_BF16(a1, bF, aF[1]);
      aG[0] = WMMA_BF16(a0, bG, aG[0]);  aG[1] = WMMA_BF16(a1, bG, aG[1]);
      aO[0] = WMMA_BF16(a0, bO, aO[0]);  aO[1] = WMMA_BF16(a1, bO, aO[1]);
    }
    int j = wv * 64 + t * 16 + n_l;       // hidden column 0..511
    float bIc = b_ih[j]        + b_hh[j];
    float bFc = b_ih[512 + j]  + b_hh[512 + j];
    float bGc = b_ih[1024 + j] + b_hh[1024 + j];
    float bOc = b_ih[1536 + j] + b_hh[1536 + j];
    for (int mt = 0; mt < 2; ++mt) {
      for (int v = 0; v < 8; ++v) {
        int m = mt * 16 + mb + v;
        size_t gi = (size_t)(m0 + m) * H_N + j;
        float ig = sigmoid_f(aI[mt][v] + bIc);
        float fg = sigmoid_f(aF[mt][v] + bFc);
        float gg = tanhf(aG[mt][v] + bGc);
        float og = sigmoid_f(aO[mt][v] + bOc);
        float cn = fg * c[gi] + ig * gg;
        float hn = og * tanhf(cn);
        o_c[gi] = cn;
        o_h[gi] = hn;
        s_hnew[m * HN_S + j] = f2bf(hn);
      }
    }
  }
  __syncthreads();

  // ---- Stage H0: out0 = relu(h_new @ ow0^T + ob0)   [32 x 512] ----------
  for (int t = 0; t < 4; ++t) {
    int nb = wv * 4 + t;
    v8f acc0 = {}, acc1 = {};
    for (int kk = 0; kk < 16; ++kk) {
      if (kk + 1 < 16) __builtin_prefetch(gbl_frag_ptr(ow0p, 16, nb, kk + 1, lane), 0, 3);
      v16bf b  = gbl_frag(ow0p, 16, nb, kk, lane);
      v16bf a0 = lds_frag(s_hnew,             HN_S, kk * 32, lane);
      v16bf a1 = lds_frag(s_hnew + 16 * HN_S, HN_S, kk * 32, lane);
      acc0 = WMMA_BF16(a0, b, acc0);
      acc1 = WMMA_BF16(a1, b, acc1);
    }
    int n = nb * 16 + n_l;
    float bias = ob0[n];
    for (int v = 0; v < 8; ++v) {
      float z0 = acc0[v] + bias, z1 = acc1[v] + bias;
      s_out0[(mb + v) * 512 + n]      = z0 > 0.0f ? z0 : 0.0f;
      s_out0[(16 + mb + v) * 512 + n] = z1 > 0.0f ? z1 : 0.0f;
    }
  }
  __syncthreads();

  // ---- Stage H1: out = sigmoid(out0 . ow1 + ob1) ------------------------
  for (int rr = 0; rr < 4; ++rr) {
    int r = wv * 4 + rr;
    float sum = 0.0f;
    for (int j = 0; j < 16; ++j) {
      int n = lane + 32 * j;
      sum += s_out0[r * 512 + n] * ow1[n];
    }
    for (int off = 16; off; off >>= 1) sum += __shfl_xor(sum, off, 32);
    if (lane == 0)
      o_out[m0 + r] = sigmoid_f(sum + ob1[0]);
  }
}

// ============================================================================
extern "C" void kernel_launch(void* const* d_in, const int* in_sizes, int n_in,
                              void* d_out, int out_size, void* d_ws, size_t ws_size,
                              hipStream_t stream) {
  const float* x    = (const float*)d_in[0];
  const float* h    = (const float*)d_in[1];
  const float* c    = (const float*)d_in[2];
  const float* aw0  = (const float*)d_in[3];
  const float* ab0  = (const float*)d_in[4];
  const float* aw1  = (const float*)d_in[5];
  const float* ab1  = (const float*)d_in[6];
  const float* w_ih = (const float*)d_in[7];
  const float* b_ih = (const float*)d_in[8];
  const float* w_hh = (const float*)d_in[9];
  const float* b_hh = (const float*)d_in[10];
  const float* ow0  = (const float*)d_in[11];
  const float* ob0  = (const float*)d_in[12];
  const float* ow1  = (const float*)d_in[13];
  const float* ob1  = (const float*)d_in[14];

  // Workspace: bf16 fragment-packed weights (~4 MiB total)
  unsigned short* ws   = (unsigned short*)d_ws;
  unsigned short* aw0p = ws;                         // 256*768
  unsigned short* aw1p = aw0p + 256 * 768;           // 256*256
  unsigned short* wihp = aw1p + 256 * 256;           // 2048*256
  unsigned short* whhp = wihp + 2048 * 256;          // 2048*512
  unsigned short* ow0p = whhp + 2048 * 512;          // 512*512

  auto pack = [&](const float* s, unsigned short* d, int N, int K) {
    int total = N * K;
    pack_weights_bf16<<<(total + 255) / 256, 256, 0, stream>>>(s, d, N, K);
  };
  pack(aw0,  aw0p, 256,  768);
  pack(aw1,  aw1p, 256,  256);
  pack(w_ih, wihp, 2048, 256);
  pack(w_hh, whhp, 2048, 512);
  pack(ow0,  ow0p, 512,  512);

  // Output layout: [out B][h_new B*512][c_new B*512][attn B*256]
  float* o_out  = (float*)d_out;
  float* o_h    = o_out + BATCH;
  float* o_c    = o_h + (size_t)BATCH * H_N;
  float* o_attn = o_c + (size_t)BATCH * H_N;

  predictor_cell_kernel<<<BATCH / MROWS, 256, 0, stream>>>(
      x, h, c, ab0, ab1, b_ih, b_hh, ob0, ow1, ob1,
      aw0p, aw1p, wihp, whhp, ow0p,
      o_out, o_h, o_c, o_attn);
}
